// FastMoEFFN_44178033607219
// MI455X (gfx1250) — matrix operand order repaired and checked by reading, not compiled, not run on hardware
//
#include <hip/hip_runtime.h>
#include <hip/hip_bf16.h>

// ---------------------------------------------------------------------------
// MoE FFN for MI455X (gfx1250).
// Pass 1: router (top-1 per batch element).
// Pass 2: pre-pack x/W1/W2 to bf16 in WMMA-fragment-native blobs in d_ws.
// Pass 3: fused FFN: h = relu(x@W1+b1); out = h@W2+b2.
//   - x blobs (128KB/WG) staged once into LDS via async-to-LDS copies
//   - W1/W2 blobs streamed from L2 with coalesced global_load_b128
//   - h tile lives in LDS; out accumulators persistent in VGPRs
// ---------------------------------------------------------------------------

#define D_H   1024
#define D_FF  4096
#define N_EXP 8
#define N_B   16
#define N_N   1024

typedef __bf16 bf16_t;
typedef bf16_t v16bf __attribute__((ext_vector_type(16)));
typedef float  v8f   __attribute__((ext_vector_type(8)));

// float -> bf16 bits (round-to-nearest-even), pure integer lowering
__device__ __forceinline__ unsigned short f2bf(float f) {
    unsigned int u = __float_as_uint(f);
    u += 0x7fffu + ((u >> 16) & 1u);
    return (unsigned short)(u >> 16);
}
__device__ __forceinline__ unsigned int pk2(float a, float b) {
    return (unsigned int)f2bf(a) | ((unsigned int)f2bf(b) << 16);
}

// --- async global->LDS copy (gfx1250 ASYNCcnt path), guarded --------------
// Probe result (round 3 diagnostic): builtin exists, 4 args, arg0 is
// non-const int4* in AS1 ("__device__"), arg1 int4* in AS3.
#if __has_builtin(__builtin_amdgcn_global_load_async_to_lds_b128)
#define HAVE_ASYNC_LDS 1
typedef int i4v __attribute__((ext_vector_type(4)));
typedef __attribute__((address_space(1))) i4v g1_i4;
typedef __attribute__((address_space(3))) i4v l3_i4;
#endif

__device__ __forceinline__ void cp16_to_lds(const unsigned short* g,
                                            unsigned short* l) {
#ifdef HAVE_ASYNC_LDS
    __builtin_amdgcn_global_load_async_to_lds_b128(
        (g1_i4*)(unsigned short*)g, (l3_i4*)l, 0, 0);
#else
    *(uint4*)l = *(const uint4*)g;
#endif
}

__device__ __forceinline__ void wait_async_copies() {
#ifdef HAVE_ASYNC_LDS
#if __has_builtin(__builtin_amdgcn_s_wait_asynccnt)
    __builtin_amdgcn_s_wait_asynccnt(0);
#else
    asm volatile("s_wait_asynccnt 0x0" ::: "memory");
#endif
#endif
}

// Fragment blob: 32 lanes x 16 bf16 (32B) contiguous per lane, 1KB total.
// Element order per lane matches the CDNA5 WMMA A/B VGPR layouts.
__device__ __forceinline__ v16bf load_blob(const unsigned short* __restrict__ p,
                                           int lane) {
    const unsigned short* q = p + (lane << 4);
    union { v16bf v; uint4 u[2]; } w;
    w.u[0] = *(const uint4*)(q);
    w.u[1] = *(const uint4*)(q + 8);
    return w.v;
}

// A-fragment from row-major LDS tile (for the h tile).
// A (16x32 bf16): lanes0-15 rows, K {0..7,16..23}; lanes16-31 K {8..15,24..31}
__device__ __forceinline__ v16bf load_frag_a_lds(const unsigned short* s,
                                                 int rowbase, int stride,
                                                 int kbase, int lane) {
    int m  = rowbase + (lane & 15);
    int hi = lane >> 4;
    const unsigned short* p = s + m * stride + kbase + hi * 8;
    union { v16bf v; uint4 q[2]; } u;
    u.q[0] = *(const uint4*)(p);
    u.q[1] = *(const uint4*)(p + 16);
    return u.v;
}

// ---------------------------------------------------------------------------
// Router: one block per batch element.
// ---------------------------------------------------------------------------
__global__ __launch_bounds__(256) void moe_router_kernel(
    const float* __restrict__ x, const float* __restrict__ Wr,
    const float* __restrict__ br, int* __restrict__ top1)
{
    __shared__ float red[256 * 8];
    const int b = blockIdx.x;
    const int t = threadIdx.x;
    const float* xb = x + (size_t)b * N_N * D_H;

    float m0 = 0.f, m1 = 0.f, m2 = 0.f, m3 = 0.f;
    for (int n = 0; n < N_N; ++n) {
        const float* p = xb + (size_t)n * D_H;
        m0 += p[t];       m1 += p[t + 256];
        m2 += p[t + 512]; m3 += p[t + 768];
    }
    float pe[8];
#pragma unroll
    for (int e = 0; e < 8; ++e) pe[e] = 0.f;
    const int   hs[4] = { t, t + 256, t + 512, t + 768 };
    const float ms[4] = { m0, m1, m2, m3 };
#pragma unroll
    for (int i = 0; i < 4; ++i)
#pragma unroll
        for (int e = 0; e < 8; ++e) pe[e] += ms[i] * Wr[hs[i] * N_EXP + e];
#pragma unroll
    for (int e = 0; e < 8; ++e) red[t * 8 + e] = pe[e];
    __syncthreads();
    for (int s = 128; s > 0; s >>= 1) {
        if (t < s) {
#pragma unroll
            for (int e = 0; e < 8; ++e) red[t * 8 + e] += red[(t + s) * 8 + e];
        }
        __syncthreads();
    }
    if (t == 0) {
        float best = -3.4e38f; int bi = 0;
#pragma unroll
        for (int e = 0; e < 8; ++e) {
            float sc = red[e] * (1.0f / (float)N_N) + br[e];
            if (sc > best) { best = sc; bi = e; }
        }
        top1[b] = bi;
    }
}

// ---------------------------------------------------------------------------
// Pack x -> A-fragment blobs: xP[b][rt(64)][kc(32)][512]
//   lane l: row = rt*16 + (l&15); elems: K = kc*32 + {hi*8+0..7, hi*8+16..23}
// Block: [32 rows][128 k] tile -> 8 blobs.  grid (8, 32, 16)
// ---------------------------------------------------------------------------
__global__ __launch_bounds__(256) void pack_x_kernel(
    const float* __restrict__ x, unsigned short* __restrict__ xP)
{
    __shared__ __align__(16) unsigned short lt[32 * 136];
    const int b  = blockIdx.z;
    const int gk = blockIdx.x;   // 128-k group
    const int gr = blockIdx.y;   // 32-row group
    const float* s = x + ((size_t)b * N_N + gr * 32) * D_H + gk * 128;
    const int tid = threadIdx.x;
    const int r = tid >> 5, k4 = tid & 31;
#pragma unroll
    for (int it = 0; it < 4; ++it) {
        int row = r + it * 8;
        float4 v = *(const float4*)(s + (size_t)row * D_H + k4 * 4);
        unsigned int* w = (unsigned int*)(lt + row * 136 + k4 * 4);
        w[0] = pk2(v.x, v.y);
        w[1] = pk2(v.z, v.w);
    }
    __syncthreads();
    const int bb  = tid >> 5;           // 8 blobs: 2 rt x 4 kc
    const int rtl = bb >> 2, kcl = bb & 3;
    const int l = tid & 31, m = l & 15, hi = l >> 4;
    const unsigned short* p0 = lt + (rtl * 16 + m) * 136 + kcl * 32 + hi * 8;
    uint4 q0 = *(const uint4*)(p0);
    uint4 q1 = *(const uint4*)(p0 + 16);
    const int rt = gr * 2 + rtl, kc = gk * 4 + kcl;
    unsigned short* o = xP + (((size_t)(b * 64 + rt) * 32 + kc) << 9) + (l << 4);
    *(uint4*)(o)     = q0;
    *(uint4*)(o + 8) = q1;
}

// ---------------------------------------------------------------------------
// Pack a [K][Nn] fp32 matrix into B-fragment blobs: dst[nt(Nn/16)][kc(K/32)][512]
//   lane l: col = nt*16 + (l&15); elems: K = kc*32 + {hi*8+0..7, hi*8+16..23}
// Block: [32 K][128 Nn] tile -> 8 blobs.  Used for W1 (K=1024,N=4096) and
// W2 (K=4096,N=1024).  grid (K/32, Nn/128, 8 experts)
// ---------------------------------------------------------------------------
__global__ __launch_bounds__(256) void pack_b_kernel(
    const float* __restrict__ src, unsigned short* __restrict__ dst,
    int K, int Nn, size_t src_estride, size_t dst_estride)
{
    __shared__ __align__(16) unsigned short lt[32 * 136];
    const int e  = blockIdx.z;
    const int gk = blockIdx.x;   // K/32
    const int gn = blockIdx.y;   // Nn/128
    const float* s = src + (size_t)e * src_estride;
    unsigned short* d = dst + (size_t)e * dst_estride;
    const int tid = threadIdx.x;
    const int kr = tid >> 5, n4 = tid & 31;
#pragma unroll
    for (int it = 0; it < 4; ++it) {
        int k = kr + it * 8;
        float4 v = *(const float4*)(s + (size_t)(gk * 32 + k) * Nn + gn * 128 + n4 * 4);
        unsigned int* w = (unsigned int*)(lt + k * 136 + n4 * 4);
        w[0] = pk2(v.x, v.y);
        w[1] = pk2(v.z, v.w);
    }
    __syncthreads();
    const int bt = tid >> 5;             // n-tile within the 128
    const int l = tid & 31, nl = l & 15, hi = l >> 4;
    unsigned int ob[8];
#pragma unroll
    for (int j = 0; j < 8; ++j) {
        int k0 = hi * 8 + 2 * j + (j & 4) * 2;
        unsigned int lo  = lt[(k0    ) * 136 + bt * 16 + nl];
        unsigned int hi2 = lt[(k0 + 1) * 136 + bt * 16 + nl];
        ob[j] = lo | (hi2 << 16);
    }
    const int nt = gn * 8 + bt;
    const int KC = K >> 5;
    unsigned short* o = d + (((size_t)nt * KC + gk) << 9) + (l << 4);
    uint4 q0; q0.x = ob[0]; q0.y = ob[1]; q0.z = ob[2]; q0.w = ob[3];
    uint4 q1; q1.x = ob[4]; q1.y = ob[5]; q1.z = ob[6]; q1.w = ob[7];
    *(uint4*)(o)     = q0;
    *(uint4*)(o + 8) = q1;
}

// ---------------------------------------------------------------------------
// Main fused FFN.  Block = 64 rows of one batch element, 8 waves = 4 rg x 2 cg.
// Wave: persistent out[16,512] = 32 C tiles (256 VGPRs).
// LDS: xL = this WG's x blobs (128KB, async-copied once) + hT (17KB).
// ---------------------------------------------------------------------------
#define XL_ELEMS (4 * 32 * 512)      // 65536 ushort = 128KB
#define HT_ELEMS (64 * 136)          // 8704  ushort = 17KB

__global__ __launch_bounds__(256) void moe_ffn_main(
    const unsigned short* __restrict__ xP, const unsigned short* __restrict__ w1P,
    const unsigned short* __restrict__ w2P, const float* __restrict__ b1,
    const float* __restrict__ b2, const int* __restrict__ top1,
    float* __restrict__ out)
{
    __shared__ __align__(16) unsigned short smem[XL_ELEMS + HT_ELEMS];
    unsigned short* xL = smem;
    unsigned short* hT = smem + XL_ELEMS;

    const int tid  = threadIdx.x;
    const int lane = tid & 31;
    const int wave = tid >> 5;
    const int rg   = wave >> 1;   // 0..3: 16-row group
    const int cg   = wave & 1;    // 0..1: 512-col group
    const int b    = blockIdx.y;
    const int rb   = blockIdx.x;  // 0..15, 64 rows each
    const int rt0  = rb * 4;
    const int r0   = rb * 64;

    const int e = top1[b];
    const unsigned short* xPb  = xP  + ((size_t)b * 64) * 32 * 512;
    const unsigned short* w1Pe = w1P + (size_t)e * 256 * 32 * 512;
    const unsigned short* w2Pe = w2P + (size_t)e * 64 * 128 * 512;
    const float* b1e = b1 + (size_t)e * D_FF;
    const float* b2e = b2 + (size_t)e * D_H;

    // ---- stage this WG's x blobs (contiguous 128KB) into LDS, once ----
    {
        const unsigned short* xsrc = xPb + ((size_t)rt0 * 32) * 512;
#pragma unroll
        for (int i = 0; i < 32; ++i) {
            int c = i * 256 + tid;            // 16B chunk index
            cp16_to_lds(xsrc + c * 8, xL + c * 8);
        }
        wait_async_copies();
    }
    __syncthreads();

    const v8f vzero = {0.f,0.f,0.f,0.f,0.f,0.f,0.f,0.f};
    v8f acc[32];
#pragma unroll
    for (int j = 0; j < 32; ++j) acc[j] = vzero;

    const int col = lane & 15, hi = lane >> 4;

    for (int fc = 0; fc < D_FF; fc += 128) {
        v8f hc[4];
#pragma unroll
        for (int j = 0; j < 4; ++j) hc[j] = vzero;

        // ---- GEMM1: h[64,128] = x[64,1024] @ W1[:,fc:fc+128] ----
        const unsigned short* aBase = xL + ((size_t)rg * 32) * 512;
        const unsigned short* bBase = w1Pe + ((size_t)((fc >> 4) + cg * 4) * 32) * 512;
        for (int kc = 0; kc < 32; ++kc) {
            v16bf a = load_blob(aBase + kc * 512, lane);
#pragma unroll
            for (int ftl = 0; ftl < 4; ++ftl) {
                v16bf bf = load_blob(bBase + (ftl * 32 + kc) * 512, lane);
                hc[ftl] = __builtin_amdgcn_wmma_f32_16x16x32_bf16(
                    false, a, false, bf, (short)0, hc[ftl], false, false);
            }
        }

        __syncthreads();   // prior GEMM2 reads of hT complete
        // ---- bias + ReLU -> hT (row-major = A2-fragment layout) ----
#pragma unroll
        for (int ftl = 0; ftl < 4; ++ftl) {
            float bias = b1e[fc + cg * 64 + ftl * 16 + col];
#pragma unroll
            for (int r = 0; r < 8; ++r) {
                int row = rg * 16 + r + hi * 8;
                float v = hc[ftl][r] + bias;
                v = v > 0.f ? v : 0.f;
                hT[row * 136 + cg * 64 + ftl * 16 + col] = f2bf(v);
            }
        }
        __syncthreads();

        // ---- GEMM2: acc += h[64,128] @ W2[fc:fc+128,:] ----
        const int fcb = fc >> 5;
#pragma unroll
        for (int fs = 0; fs < 4; ++fs) {
            v16bf a2 = load_frag_a_lds(hT, rg * 16, 136, fs * 32, lane);
            const unsigned short* b2Base =
                w2Pe + (((size_t)(cg * 32) * 128 + fcb + fs)) * 512;
#pragma unroll
            for (int j = 0; j < 32; ++j) {
                v16bf bfr = load_blob(b2Base + (size_t)j * 128 * 512, lane);
                acc[j] = __builtin_amdgcn_wmma_f32_16x16x32_bf16(
                    false, a2, false, bfr, (short)0, acc[j], false, false);
            }
        }
    }

    // ---- epilogue: + b2, fp32 store ----
#pragma unroll
    for (int j = 0; j < 32; ++j) {
        int n = (cg * 32 + j) * 16 + col;
        float bias = b2e[n];
#pragma unroll
        for (int r = 0; r < 8; ++r) {
            int row = r0 + rg * 16 + r + hi * 8;
            out[((size_t)b * N_N + row) * D_H + n] = acc[j][r] + bias;
        }
    }
}

// ---------------------------------------------------------------------------
// d_ws layout (bytes):
//   [0,64)          top1[16]
//   [1024, +32MB)   xP   16*64*32*512 bf16
//   [+, +64MB)      w1P  8*256*32*512 bf16
//   [+, +64MB)      w2P  8*64*128*512 bf16     total ~160MB
// ---------------------------------------------------------------------------
extern "C" void kernel_launch(void* const* d_in, const int* in_sizes, int n_in,
                              void* d_out, int out_size, void* d_ws, size_t ws_size,
                              hipStream_t stream) {
    const float* x  = (const float*)d_in[0];
    const float* Wr = (const float*)d_in[1];
    const float* br = (const float*)d_in[2];
    const float* W1 = (const float*)d_in[3];
    const float* b1 = (const float*)d_in[4];
    const float* W2 = (const float*)d_in[5];
    const float* b2 = (const float*)d_in[6];
    float* out = (float*)d_out;

    char* ws = (char*)d_ws;
    int* top1 = (int*)ws;
    unsigned short* xP  = (unsigned short*)(ws + 1024);
    unsigned short* w1P = (unsigned short*)(ws + 1024 + (size_t)33554432);
    unsigned short* w2P = (unsigned short*)(ws + 1024 + (size_t)33554432 + 67108864);

    moe_router_kernel<<<dim3(N_B), dim3(256), 0, stream>>>(x, Wr, br, top1);
    pack_x_kernel<<<dim3(8, 32, N_B), dim3(256), 0, stream>>>(x, xP);
    pack_b_kernel<<<dim3(32, 32, N_EXP), dim3(256), 0, stream>>>(
        W1, w1P, D_H, D_FF, (size_t)D_H * D_FF, (size_t)256 * 32 * 512);
    pack_b_kernel<<<dim3(128, 8, N_EXP), dim3(256), 0, stream>>>(
        W2, w2P, D_FF, D_H, (size_t)D_FF * D_H, (size_t)64 * 128 * 512);
    moe_ffn_main<<<dim3(16, N_B), dim3(256), 0, stream>>>(
        xP, w1P, w2P, b1, b2, top1, out);
}
